// Rationale_Generator_87600152969437
// MI455X (gfx1250) — compile-verified, matrix operand order checked by reading, not run on hardware
//
#include <hip/hip_runtime.h>
#include <hip/hip_bf16.h>

// ---------------------------------------------------------------------------
// Problem constants (from reference)
// ---------------------------------------------------------------------------
constexpr int BGR   = 128;          // graphs per batch
constexpr int NPER  = 256;          // nodes per graph
constexpr int EPER  = 4096;         // edges per graph
constexpr int FIN   = 128;          // input features
constexpr int CDIM  = 64;           // hidden channels
constexpr int KTOP  = 2048;         // edges kept per graph (cut set)
constexpr int DATTR = 4;
constexpr int NNODES = BGR * NPER;              // 32768
constexpr long long EEDGES = (long long)BGR * EPER;   // 524288
constexpr long long ECUT   = (long long)BGR * KTOP;   // 262144

// Output layout (flat float, reference return order)
constexpr long long SZ_CX   = (long long)NNODES * CDIM;   // 2097152
constexpr long long OFF_C_X    = 0;
constexpr long long OFF_C_EI   = OFF_C_X   + SZ_CX;
constexpr long long OFF_C_ATTR = OFF_C_EI  + 2 * ECUT;
constexpr long long OFF_C_W    = OFF_C_ATTR + DATTR * ECUT;
constexpr long long OFF_C_B    = OFF_C_W   + ECUT;
constexpr long long OFF_C_N    = OFF_C_B   + NNODES;
constexpr long long OFF_NC_X   = OFF_C_N   + 1;
constexpr long long OFF_NC_EI  = OFF_NC_X  + SZ_CX;
constexpr long long OFF_NC_ATTR= OFF_NC_EI + 2 * ECUT;
constexpr long long OFF_NC_W   = OFF_NC_ATTR + DATTR * ECUT;
constexpr long long OFF_NC_B   = OFF_NC_W  + ECUT;
constexpr long long OFF_NC_N   = OFF_NC_B  + NNODES;
constexpr long long OFF_SCORE  = OFF_NC_N  + 1;

// ---------------------------------------------------------------------------
// bf16 helpers + WMMA fragment types (CDNA5 wave32, 16x16x32 bf16)
// ---------------------------------------------------------------------------
typedef __attribute__((ext_vector_type(16))) __bf16 v16bf;
typedef __attribute__((ext_vector_type(8)))  float  v8f;
typedef unsigned int u32x4 __attribute__((ext_vector_type(4)));
typedef int          i32x4 __attribute__((ext_vector_type(4)));
typedef int          i32x8 __attribute__((ext_vector_type(8)));

#if defined(__has_builtin)
#if __has_builtin(__builtin_amdgcn_tensor_load_to_lds)
#define HAVE_TDM 1
#endif
#endif
#ifndef HAVE_TDM
#define HAVE_TDM 0
#endif

__device__ __forceinline__ __bf16 f2bf(float f) {
  unsigned u = __builtin_bit_cast(unsigned, f);
  unsigned r = (u + 0x7FFFu + ((u >> 16) & 1u)) >> 16;
  return __builtin_bit_cast(__bf16, (unsigned short)r);
}
__device__ __forceinline__ float bf2f(__bf16 b) {
  unsigned short s = __builtin_bit_cast(unsigned short, b);
  unsigned u = ((unsigned)s) << 16;
  return __builtin_bit_cast(float, u);
}

__device__ __forceinline__ v8f wmma_bf16(v16bf a, v16bf b, v8f c) {
  return __builtin_amdgcn_wmma_f32_16x16x32_bf16(false, a, false, b,
                                                 (short)0, c, false, false);
}

// A fragment: 16x32 bf16 tile, row-major source with stride lda.
__device__ __forceinline__ v16bf load_a_frag(const __bf16* A, int lda) {
  int lane = threadIdx.x & 31;
  const __bf16* row = A + (lane & 15) * lda;
  int kh = (lane >> 4) * 8;
  v16bf a;
#pragma unroll
  for (int j = 0; j < 4; ++j) {
    a[2 * j]     = row[kh + 2 * j];
    a[2 * j + 1] = row[kh + 2 * j + 1];
    a[8 + 2 * j]     = row[16 + kh + 2 * j];
    a[8 + 2 * j + 1] = row[16 + kh + 2 * j + 1];
  }
  return a;
}

// B fragment: 32x16 bf16 tile (K x N), row-major source with stride ldb.
__device__ __forceinline__ v16bf load_b_frag(const __bf16* B, int ldb) {
  int lane = threadIdx.x & 31;
  int n = lane & 15;
  int kh = (lane >> 4) * 8;
  v16bf b;
#pragma unroll
  for (int j = 0; j < 4; ++j) {
    b[2 * j]     = B[(kh + 2 * j) * ldb + n];
    b[2 * j + 1] = B[(kh + 2 * j + 1) * ldb + n];
    b[8 + 2 * j]     = B[(16 + kh + 2 * j) * ldb + n];
    b[8 + 2 * j + 1] = B[(16 + kh + 2 * j + 1) * ldb + n];
  }
  return b;
}

// ---------------------------------------------------------------------------
// TDM: issue one 1-row 2D tensor_load_to_lds of `elems` bf16 elements
// D# per CDNA5 ISA (group0: count/lds/global/type,
// group1: data_size=2B, dims/strides; groups 2-3 zero for <=2D).
// ---------------------------------------------------------------------------
#if HAVE_TDM
__device__ __forceinline__ void tdm_load_bf16(unsigned lds_byte_off,
                                              const __bf16* gsrc, int elems) {
  unsigned long long ga = (unsigned long long)(uintptr_t)gsrc;
  u32x4 g0;
  g0[0] = 1u;                                        // count=1, user descriptor
  g0[1] = lds_byte_off;                              // lds_addr
  g0[2] = (unsigned)(ga & 0xFFFFFFFFu);              // global_addr[31:0]
  g0[3] = (unsigned)((ga >> 32) & 0x01FFFFFFu) | (2u << 30);  // addr hi | type=2
  unsigned ue = (unsigned)elems;
  i32x8 g1;
  g1[0] = (int)(1u << 16);                           // data_size=1 (2 bytes)
  g1[1] = (int)((ue & 0xFFFFu) << 16);               // tensor_dim0 lo16
  g1[2] = (int)((ue >> 16) | (1u << 16));            // tensor_dim0 hi16 | tensor_dim1=1
  g1[3] = (int)((ue & 0xFFFFu) << 16);               // tile_dim0
  g1[4] = 1;                                         // tile_dim1=1
  g1[5] = (int)ue;                                   // tensor_dim0_stride lo32
  g1[6] = 0;
  g1[7] = 0;
  i32x4 z4 = {0, 0, 0, 0};
#if __clang_major__ >= 23
  i32x8 z8 = {0, 0, 0, 0, 0, 0, 0, 0};
  __builtin_amdgcn_tensor_load_to_lds(g0, g1, z4, z4, z8, 0);
#else
  __builtin_amdgcn_tensor_load_to_lds(g0, g1, z4, z4, 0);
#endif
}
#endif

// ---------------------------------------------------------------------------
// Kernel 1: graph conv layer 1 (one workgroup per graph, 8 waves)
// ---------------------------------------------------------------------------
__global__ __launch_bounds__(256) void conv1_kernel(
    const float* __restrict__ x, const int* __restrict__ ei,
    const float* __restrict__ W1_rel, const float* __restrict__ W1_root,
    const float* __restrict__ b1, float* __restrict__ h1) {
  extern __shared__ char smem_raw[];
  float*  agg   = (float*)smem_raw;                        // 256*128 f32
  __bf16* xs    = (__bf16*)(smem_raw + NPER * FIN * 4);    // 256*128 bf16
  __bf16* aggbf = xs + NPER * FIN;
  __bf16* wrel  = aggbf + NPER * FIN;                      // 128*64
  __bf16* wroot = wrel + FIN * CDIM;

  const int g = blockIdx.x, tid = threadIdx.x;
  const int nbase = g * NPER;
  const long long ebase = (long long)g * EPER;

  for (int i = tid; i < NPER * FIN; i += 256) {
    xs[i]  = f2bf(x[(long long)nbase * FIN + i]);
    agg[i] = 0.f;
  }
  for (int i = tid; i < FIN * CDIM; i += 256) {
    wrel[i]  = f2bf(W1_rel[i]);
    wroot[i] = f2bf(W1_root[i]);
  }
  __syncthreads();

  const int* src = ei;
  const int* dst = ei + EEDGES;
  for (int it = tid; it < EPER * 4; it += 256) {
    int e = it >> 2;
    int q = (it & 3) * 32;
    int r = src[ebase + e] - nbase;
    int c = dst[ebase + e] - nbase;
    const __bf16* xr = xs + r * FIN + q;
    float* ac = agg + c * FIN + q;
#pragma unroll
    for (int f = 0; f < 32; ++f) atomicAdd(&ac[f], bf2f(xr[f]));
  }
  __syncthreads();
  for (int i = tid; i < NPER * FIN; i += 256) aggbf[i] = f2bf(agg[i]);
  __syncthreads();

  const int wave = tid >> 5, lane = tid & 31;
  for (int t = wave; t < 16 * 4; t += 8) {
    int mt = t >> 2, nt = t & 3;
    v8f acc = {};
#pragma unroll
    for (int kt = 0; kt < 4; ++kt) {
      acc = wmma_bf16(load_a_frag(aggbf + (mt * 16) * FIN + kt * 32, FIN),
                      load_b_frag(wrel + (kt * 32) * CDIM + nt * 16, CDIM), acc);
      acc = wmma_bf16(load_a_frag(xs + (mt * 16) * FIN + kt * 32, FIN),
                      load_b_frag(wroot + (kt * 32) * CDIM + nt * 16, CDIM), acc);
    }
    int n  = nt * 16 + (lane & 15);
    int mb = mt * 16 + ((lane >> 4) * 8);
    float bias = b1[n];
#pragma unroll
    for (int v = 0; v < 8; ++v) {
      float val = acc[v] + bias;
      h1[(long long)(nbase + mb + v) * CDIM + n] = val > 0.f ? val : 0.f;
    }
  }
}

// ---------------------------------------------------------------------------
// Kernel 2: graph conv layer 2 (no relu); also emits bf16 copy of h for TDM
// ---------------------------------------------------------------------------
__global__ __launch_bounds__(256) void conv2_kernel(
    const float* __restrict__ h1, const int* __restrict__ ei,
    const float* __restrict__ W2_rel, const float* __restrict__ W2_root,
    const float* __restrict__ b2, float* __restrict__ h, __bf16* __restrict__ hbf) {
  extern __shared__ char smem_raw[];
  float*  agg   = (float*)smem_raw;                        // 256*64 f32
  __bf16* hs    = (__bf16*)(smem_raw + NPER * CDIM * 4);
  __bf16* aggbf = hs + NPER * CDIM;
  __bf16* wrel  = aggbf + NPER * CDIM;
  __bf16* wroot = wrel + CDIM * CDIM;

  const int g = blockIdx.x, tid = threadIdx.x;
  const int nbase = g * NPER;
  const long long ebase = (long long)g * EPER;

  for (int i = tid; i < NPER * CDIM; i += 256) {
    hs[i]  = f2bf(h1[(long long)nbase * CDIM + i]);
    agg[i] = 0.f;
  }
  for (int i = tid; i < CDIM * CDIM; i += 256) {
    wrel[i]  = f2bf(W2_rel[i]);
    wroot[i] = f2bf(W2_root[i]);
  }
  __syncthreads();

  const int* src = ei;
  const int* dst = ei + EEDGES;
  for (int it = tid; it < EPER * 2; it += 256) {
    int e = it >> 1;
    int q = (it & 1) * 32;
    int r = src[ebase + e] - nbase;
    int c = dst[ebase + e] - nbase;
    const __bf16* hr = hs + r * CDIM + q;
    float* ac = agg + c * CDIM + q;
#pragma unroll
    for (int f = 0; f < 32; ++f) atomicAdd(&ac[f], bf2f(hr[f]));
  }
  __syncthreads();
  for (int i = tid; i < NPER * CDIM; i += 256) aggbf[i] = f2bf(agg[i]);
  __syncthreads();

  const int wave = tid >> 5, lane = tid & 31;
  for (int t = wave; t < 16 * 4; t += 8) {
    int mt = t >> 2, nt = t & 3;
    v8f acc = {};
#pragma unroll
    for (int kt = 0; kt < 2; ++kt) {
      acc = wmma_bf16(load_a_frag(aggbf + (mt * 16) * CDIM + kt * 32, CDIM),
                      load_b_frag(wrel + (kt * 32) * CDIM + nt * 16, CDIM), acc);
      acc = wmma_bf16(load_a_frag(hs + (mt * 16) * CDIM + kt * 32, CDIM),
                      load_b_frag(wroot + (kt * 32) * CDIM + nt * 16, CDIM), acc);
    }
    int n  = nt * 16 + (lane & 15);
    int mb = mt * 16 + ((lane >> 4) * 8);
    float bias = b2[n];
#pragma unroll
    for (int v = 0; v < 8; ++v) {
      float val = acc[v] + bias;
      long long idx = (long long)(nbase + mb + v) * CDIM + n;
      h[idx] = val;
      hbf[idx] = f2bf(val);
    }
  }
}

// ---------------------------------------------------------------------------
// Prep: swizzle Wm1 into fragment-major bf16 layout so each lane's 16 B-frag
// values are 32 contiguous bytes in LDS (2x ds_load_b128 per fragment).
// Fragment (kt=k/32, nt=n/16): slot = ((kt*16+nt)*32 + lane)*16 + e
// ---------------------------------------------------------------------------
__global__ void prep_wm1_kernel(const float* __restrict__ Wm1,
                                __bf16* __restrict__ wm1bf) {
  int i = blockIdx.x * blockDim.x + threadIdx.x;    // FIN*256 = 32768
  if (i >= FIN * 256) return;
  int k = i >> 8, n = i & 255;
  int kr = k & 31;
  int kh8 = (kr < 16) ? (kr & 8) : ((kr - 16) & 8);
  int e   = (kr < 16) ? (kr & 7) : (8 + ((kr - 16) & 7));
  int lane = ((kh8 >> 3) << 4) + (n & 15);
  int frag = (k >> 5) * 16 + (n >> 4);
  wm1bf[(frag * 32 + lane) * 16 + e] = f2bf(Wm1[i]);
}

// ---------------------------------------------------------------------------
// Kernel 3: fused edge MLP (dominant GEMM). One WG per graph.
// LDS staged via TDM (tensor_load_to_lds + s_wait_tensorcnt) when available.
// ---------------------------------------------------------------------------
__global__ __launch_bounds__(256) void edge_mlp_kernel(
    const __bf16* __restrict__ hbf, const __bf16* __restrict__ wm1bf,
    const int* __restrict__ ei, const float* __restrict__ bm1,
    const float* __restrict__ Wm2, const float* __restrict__ bm2,
    float* __restrict__ scores_ws, float* __restrict__ out) {
  extern __shared__ char smem_raw[];
  __bf16* hs    = (__bf16*)smem_raw;                 // 256*64         (32 KB)
  __bf16* wm1s  = hs + NPER * CDIM;                  // swizzled Wm1   (64 KB)
  int*    srcl  = (int*)(wm1s + FIN * 256);          // 4096 ints
  int*    dstl  = srcl + EPER;
  float*  bm1s  = (float*)(dstl + EPER);             // 256 f32
  float*  wm2s  = bm1s + 256;

  const int g = blockIdx.x, tid = threadIdx.x;
  const int nbase = g * NPER;
  const long long ebase = (long long)g * EPER;

  __builtin_prefetch(wm1bf + tid * 128, 0, 3);       // global_prefetch_b8

#if HAVE_TDM
  if (tid == 0) {
    tdm_load_bf16((unsigned)(uintptr_t)hs, hbf + (long long)nbase * CDIM,
                  NPER * CDIM);
    tdm_load_bf16((unsigned)(uintptr_t)wm1s, wm1bf, FIN * 256);
    __builtin_amdgcn_s_wait_tensorcnt(0);
  }
#else
  for (int i = tid; i < NPER * CDIM; i += 256)
    hs[i] = hbf[(long long)nbase * CDIM + i];
  for (int i = tid; i < FIN * 256; i += 256)
    wm1s[i] = wm1bf[i];
#endif
  for (int i = tid; i < EPER; i += 256) {
    srcl[i] = ei[ebase + i] - nbase;
    dstl[i] = ei[EEDGES + ebase + i] - nbase;
  }
  for (int i = tid; i < 256; i += 256) {
    bm1s[i] = bm1[i];
    wm2s[i] = Wm2[i];
  }
  __syncthreads();

  const int wave = tid >> 5, lane = tid & 31;
  const int nlo = lane & 15;
  const int kh = (lane >> 4) * 8;
  const float b2v = bm2[0];

  for (int et = wave; et < EPER / 16; et += 8) {
    int e0 = et * 16;
    // A fragments (rep = [h[src] | h[dst]]) depend only on the edge tile:
    // hoisted out of the nt loop, resident in 32 VGPRs.
    v16bf afr[4];
#pragma unroll
    for (int kt = 0; kt < 4; ++kt) {
      int e = e0 + nlo;
      const __bf16* base = (kt < 2) ? (hs + srcl[e] * CDIM + kt * 32)
                                    : (hs + dstl[e] * CDIM + (kt - 2) * 32);
      v16bf a;
#pragma unroll
      for (int j = 0; j < 8; ++j) {
        a[j]     = base[kh + j];
        a[8 + j] = base[16 + kh + j];
      }
      afr[kt] = a;
    }
    float racc[8] = {0, 0, 0, 0, 0, 0, 0, 0};
    for (int nt = 0; nt < 16; ++nt) {                // 256 hidden cols / 16
      v8f acc = {};
#pragma unroll
      for (int kt = 0; kt < 4; ++kt) {
        const __bf16* fb = wm1s + ((kt * 16 + nt) * 32 + lane) * 16;
        acc = wmma_bf16(afr[kt], *(const v16bf*)fb, acc);
      }
      int n = nt * 16 + nlo;
      float bias = bm1s[n];
      float w2 = wm2s[n];
#pragma unroll
      for (int v = 0; v < 8; ++v) {
        float hv = acc[v] + bias;
        hv = hv > 0.f ? hv : 0.f;                    // relu
        racc[v] += hv * w2;                          // fused 256->1 projection
      }
    }
#pragma unroll
    for (int v = 0; v < 8; ++v) {
      float s = racc[v];
      s += __shfl_xor(s, 1, 32);
      s += __shfl_xor(s, 2, 32);
      s += __shfl_xor(s, 4, 32);
      s += __shfl_xor(s, 8, 32);
      racc[v] = s;
    }
    if (nlo == 0) {
      int mb = (lane >> 4) * 8;
#pragma unroll
      for (int v = 0; v < 8; ++v) {
        long long eg = ebase + e0 + mb + v;
        float sc = racc[v] + b2v;
        scores_ws[eg] = sc;
        out[OFF_SCORE + eg] = sc;
      }
    }
  }
}

// ---------------------------------------------------------------------------
// Kernel 4: per-graph descending bitonic sort of 4096 scores + partition
// ---------------------------------------------------------------------------
__global__ __launch_bounds__(1024) void sort_partition_kernel(
    const float* __restrict__ scores, const int* __restrict__ ei,
    const float* __restrict__ edge_attr, float* __restrict__ out,
    int* __restrict__ c_src, int* __restrict__ c_dst,
    int* __restrict__ nc_src, int* __restrict__ nc_dst) {
  __shared__ float key[EPER];
  __shared__ int   val[EPER];
  const int g = blockIdx.x, tid = threadIdx.x;
  const long long ebase = (long long)g * EPER;

  for (int i = tid; i < EPER; i += 1024) { key[i] = scores[ebase + i]; val[i] = i; }
  __syncthreads();

  for (int k = 2; k <= EPER; k <<= 1) {
    for (int j = k >> 1; j > 0; j >>= 1) {
      for (int i = tid; i < EPER; i += 1024) {
        int ixj = i ^ j;
        if (ixj > i) {
          float ki = key[i], kj = key[ixj];
          int   vi = val[i], vj = val[ixj];
          bool after = (ki < kj) || (ki == kj && vi > vj);
          bool desc = ((i & k) == 0);
          if (after == desc) {
            key[i] = kj; key[ixj] = ki;
            val[i] = vj; val[ixj] = vi;
          }
        }
      }
      __syncthreads();
    }
  }

  const int* src = ei;
  const int* dst = ei + EEDGES;
  for (int i = tid; i < EPER; i += 1024) {
    int le = val[i];
    long long ge = ebase + le;
    int s = src[ge], d = dst[ge];
    if (i < KTOP) {
      long long o = (long long)g * KTOP + i;
      out[OFF_C_W + o] = key[i];
#pragma unroll
      for (int a = 0; a < DATTR; ++a)
        out[OFF_C_ATTR + o * DATTR + a] = edge_attr[ge * DATTR + a];
      c_src[o] = s; c_dst[o] = d;
    } else {
      long long o = (long long)g * KTOP + (i - KTOP);
      out[OFF_NC_W + o] = -key[i];
#pragma unroll
      for (int a = 0; a < DATTR; ++a)
        out[OFF_NC_ATTR + o * DATTR + a] = edge_attr[ge * DATTR + a];
      nc_src[o] = s; nc_dst[o] = d;
    }
  }
}

// ---------------------------------------------------------------------------
// Relabel machinery: init, mask, scan, scatter
// ---------------------------------------------------------------------------
__global__ void init_out_kernel(float* __restrict__ out,
                                int* __restrict__ mask_c, int* __restrict__ mask_nc) {
  long long i = (long long)blockIdx.x * blockDim.x + threadIdx.x;
  long long stride = (long long)gridDim.x * blockDim.x;
  for (long long t = i; t < SZ_CX; t += stride) {
    out[OFF_C_X + t] = 0.f;
    out[OFF_NC_X + t] = 0.f;
  }
  for (long long t = i; t < NNODES; t += stride) {
    out[OFF_C_B + t] = -1.f;
    out[OFF_NC_B + t] = -1.f;
    mask_c[t] = 0;
    mask_nc[t] = 0;
  }
}

__global__ void mask_set_kernel(const int* __restrict__ c_src, const int* __restrict__ c_dst,
                                const int* __restrict__ nc_src, const int* __restrict__ nc_dst,
                                int* __restrict__ mask_c, int* __restrict__ mask_nc) {
  int i = blockIdx.x * blockDim.x + threadIdx.x;
  if (i < (int)ECUT) {
    mask_c[c_src[i]] = 1;
    mask_c[c_dst[i]] = 1;
    mask_nc[nc_src[i]] = 1;
    mask_nc[nc_dst[i]] = 1;
  }
}

__global__ __launch_bounds__(1024) void scan_kernel(
    const int* __restrict__ mask_c, const int* __restrict__ mask_nc,
    int* __restrict__ pos_c, int* __restrict__ pos_nc, float* __restrict__ out) {
  const int* mask = blockIdx.x ? mask_nc : mask_c;
  int* pos = blockIdx.x ? pos_nc : pos_c;
  const long long cntOff = blockIdx.x ? OFF_NC_N : OFF_C_N;
  __shared__ int part[1024];
  const int tid = threadIdx.x;
  const int base = tid * 32;
  int s = 0;
  for (int k = 0; k < 32; ++k) s += mask[base + k];
  part[tid] = s;
  __syncthreads();
  for (int off = 1; off < 1024; off <<= 1) {
    int v = (tid >= off) ? part[tid - off] : 0;
    __syncthreads();
    part[tid] += v;
    __syncthreads();
  }
  int run = (tid == 0) ? 0 : part[tid - 1];
  for (int k = 0; k < 32; ++k) { pos[base + k] = run; run += mask[base + k]; }
  if (tid == 1023) out[cntOff] = (float)part[1023];
}

__global__ void scatter_nodes_kernel(
    const float* __restrict__ h, const int* __restrict__ batch,
    const int* __restrict__ mask_c, const int* __restrict__ pos_c,
    const int* __restrict__ mask_nc, const int* __restrict__ pos_nc,
    float* __restrict__ out) {
  int i = blockIdx.x * blockDim.x + threadIdx.x;
  if (i >= NNODES) return;
  float bval = (float)batch[i];
  if (mask_c[i]) {
    int p = pos_c[i];
    for (int f = 0; f < CDIM; ++f)
      out[OFF_C_X + (long long)p * CDIM + f] = h[(long long)i * CDIM + f];
    out[OFF_C_B + p] = bval;
  }
  if (mask_nc[i]) {
    int p = pos_nc[i];
    for (int f = 0; f < CDIM; ++f)
      out[OFF_NC_X + (long long)p * CDIM + f] = h[(long long)i * CDIM + f];
    out[OFF_NC_B + p] = bval;
  }
}

__global__ void relabel_edges_kernel(
    const int* __restrict__ c_src, const int* __restrict__ c_dst,
    const int* __restrict__ nc_src, const int* __restrict__ nc_dst,
    const int* __restrict__ mask_c, const int* __restrict__ pos_c,
    const int* __restrict__ mask_nc, const int* __restrict__ pos_nc,
    float* __restrict__ out) {
  int i = blockIdx.x * blockDim.x + threadIdx.x;
  if (i >= (int)ECUT) return;
  int s = c_src[i], d = c_dst[i];
  out[OFF_C_EI + i]        = (float)(mask_c[s] ? pos_c[s] : -1);
  out[OFF_C_EI + ECUT + i] = (float)(mask_c[d] ? pos_c[d] : -1);
  s = nc_src[i]; d = nc_dst[i];
  out[OFF_NC_EI + i]        = (float)(mask_nc[s] ? pos_nc[s] : -1);
  out[OFF_NC_EI + ECUT + i] = (float)(mask_nc[d] ? pos_nc[d] : -1);
}

// ---------------------------------------------------------------------------
// Launch
// ---------------------------------------------------------------------------
extern "C" void kernel_launch(void* const* d_in, const int* in_sizes, int n_in,
                              void* d_out, int out_size, void* d_ws, size_t ws_size,
                              hipStream_t stream) {
  const float* x        = (const float*)d_in[0];
  const int*   ei       = (const int*)d_in[1];
  const int*   batch    = (const int*)d_in[2];
  const float* eattr    = (const float*)d_in[3];
  const float* W1_rel   = (const float*)d_in[4];
  const float* W1_root  = (const float*)d_in[5];
  const float* b1       = (const float*)d_in[6];
  const float* W2_rel   = (const float*)d_in[7];
  const float* W2_root  = (const float*)d_in[8];
  const float* b2       = (const float*)d_in[9];
  const float* Wm1      = (const float*)d_in[10];
  const float* bm1      = (const float*)d_in[11];
  const float* Wm2      = (const float*)d_in[12];
  const float* bm2      = (const float*)d_in[13];
  float* out = (float*)d_out;

  // workspace carve-up (~28 MB)
  char* ws = (char*)d_ws;
  float*  h1      = (float*)(ws);                    // 8 MB
  float*  h       = (float*)(ws + (8LL << 20));      // 8 MB
  float*  scores  = (float*)(ws + (16LL << 20));     // 2 MB
  int*    c_src   = (int*)(ws + (18LL << 20));       // 1 MB each
  int*    c_dst   = (int*)(ws + (19LL << 20));
  int*    nc_src  = (int*)(ws + (20LL << 20));
  int*    nc_dst  = (int*)(ws + (21LL << 20));
  int*    mask_c  = (int*)(ws + (22LL << 20));
  int*    mask_nc = mask_c + NNODES;
  int*    pos_c   = mask_nc + NNODES;
  int*    pos_nc  = pos_c + NNODES;
  __bf16* hbf     = (__bf16*)(ws + (23LL << 20));    // 4 MB
  __bf16* wm1bf   = (__bf16*)(ws + (27LL << 20));    // 64 KB (swizzled)

  const int SMEM_CONV1 = NPER * FIN * 4 + NPER * FIN * 2 * 2 + FIN * CDIM * 2 * 2;    // 294912
  const int SMEM_CONV2 = NPER * CDIM * 4 + NPER * CDIM * 2 * 2 + CDIM * CDIM * 2 * 2; // 147456
  const int SMEM_MLP   = NPER * CDIM * 2 + FIN * 256 * 2 + EPER * 4 * 2 + 2048;       // 133120
  (void)hipFuncSetAttribute((const void*)conv1_kernel,
      hipFuncAttributeMaxDynamicSharedMemorySize, SMEM_CONV1);
  (void)hipFuncSetAttribute((const void*)conv2_kernel,
      hipFuncAttributeMaxDynamicSharedMemorySize, SMEM_CONV2);
  (void)hipFuncSetAttribute((const void*)edge_mlp_kernel,
      hipFuncAttributeMaxDynamicSharedMemorySize, SMEM_MLP);

  init_out_kernel<<<512, 256, 0, stream>>>(out, mask_c, mask_nc);
  prep_wm1_kernel<<<(FIN * 256) / 256, 256, 0, stream>>>(Wm1, wm1bf);
  conv1_kernel<<<BGR, 256, SMEM_CONV1, stream>>>(x, ei, W1_rel, W1_root, b1, h1);
  conv2_kernel<<<BGR, 256, SMEM_CONV2, stream>>>(h1, ei, W2_rel, W2_root, b2, h, hbf);
  edge_mlp_kernel<<<BGR, 256, SMEM_MLP, stream>>>(hbf, wm1bf, ei, bm1, Wm2, bm2,
                                                  scores, out);
  sort_partition_kernel<<<BGR, 1024, 0, stream>>>(scores, ei, eattr, out,
                                                  c_src, c_dst, nc_src, nc_dst);
  mask_set_kernel<<<(int)(ECUT / 256), 256, 0, stream>>>(c_src, c_dst, nc_src, nc_dst,
                                                         mask_c, mask_nc);
  scan_kernel<<<2, 1024, 0, stream>>>(mask_c, mask_nc, pos_c, pos_nc, out);
  scatter_nodes_kernel<<<NNODES / 256, 256, 0, stream>>>(h, batch, mask_c, pos_c,
                                                         mask_nc, pos_nc, out);
  relabel_edges_kernel<<<(int)(ECUT / 256), 256, 0, stream>>>(c_src, c_dst, nc_src, nc_dst,
                                                              mask_c, pos_c, mask_nc, pos_nc, out);
}
